// DemiurgicAttention_57655640982132
// MI455X (gfx1250) — matrix operand order repaired and checked by reading, not compiled, not run on hardware
//
#include <hip/hip_runtime.h>
#include <hip/hip_bf16.h>

// ---------------------------------------------------------------------------
// GQA attention layer for MI455X (gfx1250, wave32, WMMA).
//   B=2, S=2048, H=2048, NH=32, NKV=8, HD=64, N_REP=4, rope theta 1e4.
// Pipeline: cast->bf16, QKV NT-GEMMs (wmma bf16, TDM tensor_load_to_lds
// double buffered), RoPE, flash attention (wmma bf16 QK^T/PV + online f32
// softmax, async-to-LDS K staging), O-proj NT-GEMM (f32 out).
// ---------------------------------------------------------------------------

#define DINL __device__ __forceinline__

typedef __attribute__((ext_vector_type(16))) __bf16 v16bf;
typedef __attribute__((ext_vector_type(8)))  __bf16 v8bf;
typedef __attribute__((ext_vector_type(8)))  float  v8f;
typedef __attribute__((ext_vector_type(4)))  unsigned int u32x4;
typedef __attribute__((ext_vector_type(8)))  int i32x8;
typedef __attribute__((ext_vector_type(4)))  int i32x4;

constexpr int B_  = 2;
constexpr int S_  = 2048;
constexpr int H_  = 2048;
constexpr int NH_ = 32;
constexpr int NKV_= 8;
constexpr int HD_ = 64;
constexpr int MT_ = B_ * S_;        // 4096 tokens
constexpr int QD_ = NH_ * HD_;      // 2048
constexpr int KD_ = NKV_ * HD_;     // 512

DINL unsigned short f2bf(float f) {
  unsigned u = __float_as_uint(f);
  u += 0x7fffu + ((u >> 16) & 1u);          // round-to-nearest-even
  return (unsigned short)(u >> 16);
}
DINL float bf2f(unsigned short h) { return __uint_as_float(((unsigned)h) << 16); }

// --- CDNA5 async global->LDS copy (ASYNCcnt) -------------------------------
DINL void async_ld_b128(unsigned short* lds_dst, const unsigned short* gsrc) {
  unsigned lds_off = (unsigned)(unsigned long long)lds_dst;  // low 32b = LDS byte offset
  asm volatile("global_load_async_to_lds_b128 %0, %1, off"
               :: "v"(lds_off), "v"(gsrc)
               : "memory");
}
DINL void wait_async0() {
  asm volatile("s_wait_asynccnt 0" ::: "memory");
}

// --- CDNA5 Tensor Data Mover: 2D bf16 tile -> LDS (TENSORcnt) --------------
// Loads a (tile_d1 x tile_d0) bf16 tile whose rows are `stride` elements apart
// in global memory into LDS, inserting 4 DWORDs of padding every 16 DWORDs
// (row = 32 bf16 = 16 DWORDs) => effective LDS row stride = 40 ushorts.
DINL void tdm_load_tile_2d(unsigned short* lds_dst, const unsigned short* gsrc,
                           int tensor_d0, int tensor_d1,
                           int tile_d0, int tile_d1, int stride) {
  unsigned lds_off = (unsigned)(unsigned long long)lds_dst;
  unsigned long long ga = (unsigned long long)gsrc;
  u32x4 g0;
  g0[0] = 1u;                                          // count=1, user descriptor
  g0[1] = lds_off;                                     // lds_addr
  g0[2] = (unsigned)(ga & 0xffffffffu);                // global_addr[31:0]
  g0[3] = (unsigned)((ga >> 32) & 0x01ffffffu)         // global_addr[56:32]
        | 0x80000000u;                                 // type=2 ("image")
  i32x8 g1;
  g1[0] = (int)((1u << 16)                             // data_size = 2 bytes
              | (1u << 20)                             // pad_enable
              | (3u << 22)                             // pad_interval: 16 DWORDs
              | (3u << 25));                           // pad_amount: 4 DWORDs
  g1[1] = (int)(((unsigned)tensor_d0 & 0xffffu) << 16);            // dim0[15:0]
  g1[2] = (int)((((unsigned)tensor_d0 >> 16) & 0xffffu)
              | (((unsigned)tensor_d1 & 0xffffu) << 16));          // dim0[31:16]|dim1[15:0]
  g1[3] = (int)((((unsigned)tensor_d1 >> 16) & 0xffffu)
              | ((unsigned)tile_d0 << 16));                        // dim1[31:16]|tile0
  g1[4] = (int)((unsigned)tile_d1 & 0xffffu);                      // tile1 (tile2=0)
  g1[5] = stride;                                                  // dim0_stride[31:0]
  g1[6] = 0;                                                       // stride hi / dim1_stride lo
  g1[7] = 0;
  i32x4 g2; g2[0] = 1; g2[1] = 1; g2[2] = 0; g2[3] = 0;  // tensor_dim2/3=1, tile_dim3=0
  i32x4 g3; g3[0] = 0; g3[1] = 0; g3[2] = 1; g3[3] = 0;  // tensor_dim4=1, tile_dim4=0
  i32x8 g4 = i32x8{0,0,0,0,0,0,0,0};                     // (toolchain 6-arg form)
  __builtin_amdgcn_tensor_load_to_lds(g0, g1, g2, g3, g4, 0);
}
DINL void wait_tensor0() { __builtin_amdgcn_s_wait_tensorcnt(0); }

// --- WMMA fragment loaders (CDNA5 ISA 7.12.2 layouts) ----------------------
// A (16x32 bf16, MxK): lane holds row M=(lane&15); K = {kb..kb+7, kb+16..kb+23},
// kb = (lane>>4)*8. Two 16B loads.
DINL v16bf ld_a16x32(const unsigned short* base, int ld) {
  int lane = (int)(threadIdx.x & 31u);
  const __bf16* p = (const __bf16*)base + (size_t)(lane & 15) * ld + ((lane >> 4) << 3);
  v8bf lo = *(const v8bf*)p;
  v8bf hi = *(const v8bf*)(p + 16);
  return __builtin_shufflevector(lo, hi, 0,1,2,3,4,5,6,7,8,9,10,11,12,13,14,15);
}
// B (32x16 bf16, KxN): lane holds col N=(lane&15); K = {kb..kb+7, kb+16..kb+23},
// kb = (lane>>4)*8.
DINL v16bf ld_b32x16(const unsigned short* base, int ld) {
  int lane = (int)(threadIdx.x & 31u);
  const __bf16* p = (const __bf16*)base + (size_t)(lane & 15) * ld + ((lane >> 4) << 3);
  v8bf lo = *(const v8bf*)p;
  v8bf hi = *(const v8bf*)(p + 16);
  return __builtin_shufflevector(lo, hi, 0,1,2,3,4,5,6,7,8,9,10,11,12,13,14,15);
}
DINL v8f wmma_bf16(v16bf a, v16bf b, v8f c) {
  return __builtin_amdgcn_wmma_f32_16x16x32_bf16(false, a, false, b, (short)0, c, false, false);
}

// --- f32 -> bf16 cast ------------------------------------------------------
__global__ __launch_bounds__(256) void cast_bf16_kernel(const float* __restrict__ in,
                                                        unsigned short* __restrict__ out,
                                                        int n) {
  int i = (int)(blockIdx.x * 256 + threadIdx.x) * 4;
  if (i + 3 < n) {
    float4 f = *(const float4*)(in + i);
    out[i + 0] = f2bf(f.x); out[i + 1] = f2bf(f.y);
    out[i + 2] = f2bf(f.z); out[i + 3] = f2bf(f.w);
  } else {
    for (int k = i; k < n; ++k) out[k] = f2bf(in[k]);
  }
}

// --- NT GEMM: C[M,N] = A[M,K] * W[N,K]^T, bf16 in, f32 accum ---------------
// Block tile 128x128x32, 256 threads = 8 waves (2x4). Tiles DMA'd to LDS by
// the Tensor Data Mover (wave 0 issues 2 descriptors per step), double
// buffered; one barrier per K step.
template <bool F32OUT>
__global__ __launch_bounds__(256) void gemm_nt_kernel(const unsigned short* __restrict__ A,
                                                      const unsigned short* __restrict__ W,
                                                      unsigned short* __restrict__ Cb,
                                                      float* __restrict__ Cf,
                                                      int M, int N, int K) {
  __shared__ __align__(16) unsigned short ldsA[2][128 * 40];  // TDM pads rows to 40
  __shared__ __align__(16) unsigned short ldsB[2][128 * 40];
  const int tid  = (int)threadIdx.x;
  const int lane = tid & 31;
  const int wave = tid >> 5;
  const int wm   = wave >> 2;        // 0..1
  const int wn   = wave & 3;         // 0..3
  const int m0   = (int)blockIdx.y * 128;
  const int n0   = (int)blockIdx.x * 128;

  v8f acc[4][2];
#pragma unroll
  for (int i = 0; i < 4; ++i)
#pragma unroll
    for (int j = 0; j < 2; ++j) acc[i][j] = v8f{0.f,0.f,0.f,0.f,0.f,0.f,0.f,0.f};

  const int nk = K >> 5;
  if (wave == 0) {
    tdm_load_tile_2d(ldsA[0], A + (size_t)m0 * K, K, M, 32, 128, K);
    tdm_load_tile_2d(ldsB[0], W + (size_t)n0 * K, K, N, 32, 128, K);
  }

  for (int s = 0; s < nk; ++s) {
    if (wave == 0) wait_tensor0();   // this buffer's DMA done
    __syncthreads();                 // visible to all waves; prev reads done
    const int cur = s & 1;
    if (s + 1 < nk && wave == 0) {   // kick next tile's DMA while computing
      tdm_load_tile_2d(ldsA[cur ^ 1], A + (size_t)m0 * K + (s + 1) * 32, K, M, 32, 128, K);
      tdm_load_tile_2d(ldsB[cur ^ 1], W + (size_t)n0 * K + (s + 1) * 32, K, N, 32, 128, K);
    }
    v16bf af[4], bfr[2];
#pragma unroll
    for (int i = 0; i < 4; ++i) af[i]  = ld_a16x32(&ldsA[cur][(wm * 64 + i * 16) * 40], 40);
#pragma unroll
    for (int j = 0; j < 2; ++j) bfr[j] = ld_b32x16(&ldsB[cur][(wn * 32 + j * 16) * 40], 40);
#pragma unroll
    for (int i = 0; i < 4; ++i)
#pragma unroll
      for (int j = 0; j < 2; ++j) acc[i][j] = wmma_bf16(af[i], bfr[j], acc[i][j]);
  }

  const int rl = lane & 15;
  const int rh = (lane >> 4) << 3;
#pragma unroll
  for (int i = 0; i < 4; ++i)
#pragma unroll
    for (int j = 0; j < 2; ++j)
#pragma unroll
      for (int jj = 0; jj < 8; ++jj) {
        size_t r = (size_t)(m0 + wm * 64 + i * 16 + jj + rh);
        size_t c = (size_t)(n0 + wn * 32 + j * 16 + rl);
        float v = acc[i][j][jj];
        if (F32OUT) Cf[r * (size_t)N + c] = v;
        else        Cb[r * (size_t)N + c] = f2bf(v);
      }
}

// --- RoPE (in place, bf16) -------------------------------------------------
__global__ __launch_bounds__(256) void rope_kernel(unsigned short* __restrict__ X,
                                                   int nheads, int total) {
  int idx = (int)(blockIdx.x * 256 + threadIdx.x);
  if (idx >= total) return;
  int d   = idx & 31;
  int t   = idx >> 5;
  int hh  = t % nheads;
  int tok = t / nheads;
  int pos = tok & (S_ - 1);
  float inv = powf(10000.f, -(float)(2 * d) / (float)HD_);
  float theta = (float)pos * inv;
  float sn, cs;
  sincosf(theta, &sn, &cs);
  unsigned short* p = X + (size_t)tok * (nheads * HD_) + hh * HD_ + d;
  float x0 = bf2f(p[0]);
  float x1 = bf2f(p[32]);
  p[0]  = f2bf(x0 * cs - x1 * sn);
  p[32] = f2bf(x1 * cs + x0 * sn);
}

// --- Flash attention (bf16 wmma, f32 online softmax) -----------------------
// grid = (S/64, NH, B), block = 128 (4 waves). Wave w owns q rows
// [qt*64 + w*16, +16). kv tiles of 64 staged in LDS (K async, V transposed).
__global__ __launch_bounds__(128) void attn_kernel(const unsigned short* __restrict__ Q,
                                                   const unsigned short* __restrict__ K,
                                                   const unsigned short* __restrict__ V,
                                                   unsigned short* __restrict__ AO) {
  __shared__ __align__(16) unsigned short ldsK[64 * 64];      // [kv][hd]
  __shared__ __align__(16) unsigned short ldsVt[64 * 64];     // [hd][kv]
  __shared__ __align__(16) unsigned short ldsP[4 * 16 * 64];  // per-wave [q][kv]

  const int tid  = (int)threadIdx.x;
  const int lane = tid & 31;
  const int wave = tid >> 5;
  const int qt   = (int)blockIdx.x;
  const int h    = (int)blockIdx.y;
  const int b    = (int)blockIdx.z;
  const int hk   = h >> 2;                        // GQA: 4 q heads per kv head

  const unsigned short* qbase =
      Q + ((size_t)(b * S_ + qt * 64 + wave * 16)) * QD_ + h * HD_;
  v16bf qa0 = ld_a16x32(qbase, QD_);          // hd 0..31
  v16bf qa1 = ld_a16x32(qbase + 32, QD_);     // hd 32..63

  v8f o[4];
#pragma unroll
  for (int nt = 0; nt < 4; ++nt) o[nt] = v8f{0.f,0.f,0.f,0.f,0.f,0.f,0.f,0.f};
  float mrow[8], lrow[8];
#pragma unroll
  for (int jj = 0; jj < 8; ++jj) { mrow[jj] = -3.0e38f; lrow[jj] = 0.f; }

  const float scale = 0.125f;                  // 1/sqrt(64)
  const int rl = lane & 15;
  const int rh = (lane >> 4) << 3;

  for (int kt = 0; kt <= qt; ++kt) {
    __syncthreads();
    // Stage K tile [64 kv][64 hd] via async global->LDS
#pragma unroll
    for (int p = 0; p < 4; ++p) {
      int c = tid + p * 128;
      int row = c >> 3, kc = c & 7;
      async_ld_b128(&ldsK[row * 64 + kc * 8],
                    &K[((size_t)(b * S_ + kt * 64 + row)) * KD_ + hk * HD_ + kc * 8]);
    }
    // Stage V tile transposed [64 hd][64 kv]
#pragma unroll
    for (int p = 0; p < 8; ++p) {
      int c = tid + p * 128;
      int row = c >> 4, dc = c & 15;
      uint2 u = *(const uint2*)&V[((size_t)(b * S_ + kt * 64 + row)) * KD_ + hk * HD_ + dc * 4];
      ldsVt[(dc * 4 + 0) * 64 + row] = (unsigned short)(u.x & 0xffffu);
      ldsVt[(dc * 4 + 1) * 64 + row] = (unsigned short)(u.x >> 16);
      ldsVt[(dc * 4 + 2) * 64 + row] = (unsigned short)(u.y & 0xffffu);
      ldsVt[(dc * 4 + 3) * 64 + row] = (unsigned short)(u.y >> 16);
    }
    wait_async0();
    __syncthreads();

    // S = Q K^T : 4 n-tiles (16 kv each) x 2 K-chunks (hd 32 each)
    v8f sc[4];
#pragma unroll
    for (int nt = 0; nt < 4; ++nt) {
      v16bf kb0 = ld_b32x16(&ldsK[nt * 16 * 64], 64);
      v16bf kb1 = ld_b32x16(&ldsK[nt * 16 * 64 + 32], 64);
      v8f z = v8f{0.f,0.f,0.f,0.f,0.f,0.f,0.f,0.f};
      sc[nt] = wmma_bf16(qa0, kb0, z);
      sc[nt] = wmma_bf16(qa1, kb1, sc[nt]);
    }

    // scale + causal mask (diagonal tile only; uniform branch)
    const int rowbase = qt * 64 + wave * 16 + rh;
#pragma unroll
    for (int nt = 0; nt < 4; ++nt)
#pragma unroll
      for (int jj = 0; jj < 8; ++jj) {
        float sv = sc[nt][jj] * scale;
        if (kt == qt) {
          int col = kt * 64 + nt * 16 + rl;
          if (col > rowbase + jj) sv = -1.0e30f;
        }
        sc[nt][jj] = sv;
      }

    // online softmax (row stats replicated across the 16 lanes of a row)
    float rmax[8], rsum[8], alpha[8];
#pragma unroll
    for (int jj = 0; jj < 8; ++jj) {
      float m = sc[0][jj];
      m = fmaxf(m, sc[1][jj]); m = fmaxf(m, sc[2][jj]); m = fmaxf(m, sc[3][jj]);
      rmax[jj] = m;
    }
#pragma unroll
    for (int mk = 8; mk >= 1; mk >>= 1)
#pragma unroll
      for (int jj = 0; jj < 8; ++jj)
        rmax[jj] = fmaxf(rmax[jj], __shfl_xor(rmax[jj], mk, 32));
#pragma unroll
    for (int jj = 0; jj < 8; ++jj) {
      float mnew = fmaxf(mrow[jj], rmax[jj]);
      alpha[jj] = __expf(mrow[jj] - mnew);
      mrow[jj] = mnew;
      rsum[jj] = 0.f;
    }
#pragma unroll
    for (int nt = 0; nt < 4; ++nt)
#pragma unroll
      for (int jj = 0; jj < 8; ++jj) {
        float pv = __expf(sc[nt][jj] - mrow[jj]);
        sc[nt][jj] = pv;
        rsum[jj] += pv;
      }
#pragma unroll
    for (int mk = 8; mk >= 1; mk >>= 1)
#pragma unroll
      for (int jj = 0; jj < 8; ++jj)
        rsum[jj] += __shfl_xor(rsum[jj], mk, 32);
#pragma unroll
    for (int jj = 0; jj < 8; ++jj) lrow[jj] = lrow[jj] * alpha[jj] + rsum[jj];
#pragma unroll
    for (int nt = 0; nt < 4; ++nt)
#pragma unroll
      for (int jj = 0; jj < 8; ++jj) o[nt][jj] *= alpha[jj];

    // P: C-layout regs -> LDS (bf16) -> reload in A layout (per-wave region;
    // LDS ops of a wave are in order, so no barrier needed).
    unsigned short* pp = &ldsP[wave * 16 * 64];
#pragma unroll
    for (int nt = 0; nt < 4; ++nt)
#pragma unroll
      for (int jj = 0; jj < 8; ++jj)
        pp[(jj + rh) * 64 + nt * 16 + rl] = f2bf(sc[nt][jj]);
    v16bf pa0 = ld_a16x32(pp, 64);        // kv 0..31
    v16bf pa1 = ld_a16x32(pp + 32, 64);   // kv 32..63

    // O += P V : B-frags from transposed V tile
#pragma unroll
    for (int nt = 0; nt < 4; ++nt) {
      v16bf vb0 = ld_b32x16(&ldsVt[nt * 16 * 64], 64);
      v16bf vb1 = ld_b32x16(&ldsVt[nt * 16 * 64 + 32], 64);
      o[nt] = wmma_bf16(pa0, vb0, o[nt]);
      o[nt] = wmma_bf16(pa1, vb1, o[nt]);
    }
  }

  // epilogue: normalize and store bf16 attention output [tok][h*64+d]
#pragma unroll
  for (int jj = 0; jj < 8; ++jj) lrow[jj] = 1.f / lrow[jj];
#pragma unroll
  for (int nt = 0; nt < 4; ++nt)
#pragma unroll
    for (int jj = 0; jj < 8; ++jj) {
      size_t tok = (size_t)(b * S_ + qt * 64 + wave * 16 + jj + rh);
      AO[tok * QD_ + h * HD_ + nt * 16 + rl] = f2bf(o[nt][jj] * lrow[jj]);
    }
}

// --- host-side orchestration ----------------------------------------------
extern "C" void kernel_launch(void* const* d_in, const int* in_sizes, int n_in,
                              void* d_out, int out_size, void* d_ws, size_t ws_size,
                              hipStream_t stream) {
  const float* X  = (const float*)d_in[0];
  const float* Wq = (const float*)d_in[1];
  const float* Wk = (const float*)d_in[2];
  const float* Wv = (const float*)d_in[3];
  const float* Wo = (const float*)d_in[4];
  float* out = (float*)d_out;

  // workspace layout (bf16 as ushort)
  char* ws = (char*)d_ws;
  unsigned short* Xb  = (unsigned short*)(ws);
  unsigned short* Wqb = (unsigned short*)(ws + 16777216);
  unsigned short* Wkb = (unsigned short*)(ws + 25165824);
  unsigned short* Wvb = (unsigned short*)(ws + 27262976);
  unsigned short* Wob = (unsigned short*)(ws + 29360128);
  unsigned short* Qb  = (unsigned short*)(ws + 37748736);
  unsigned short* Kb  = (unsigned short*)(ws + 54525952);
  unsigned short* Vb  = (unsigned short*)(ws + 58720256);
  unsigned short* AOb = (unsigned short*)(ws + 62914560);

  const int nX  = MT_ * H_;       // 8388608
  const int nWq = QD_ * H_;       // 4194304
  const int nWk = KD_ * H_;       // 1048576
  const int nWo = H_ * QD_;       // 4194304
  auto blocks4 = [](int n) { return (n / 4 + 255) / 256; };
  cast_bf16_kernel<<<blocks4(nX),  256, 0, stream>>>(X,  Xb,  nX);
  cast_bf16_kernel<<<blocks4(nWq), 256, 0, stream>>>(Wq, Wqb, nWq);
  cast_bf16_kernel<<<blocks4(nWk), 256, 0, stream>>>(Wk, Wkb, nWk);
  cast_bf16_kernel<<<blocks4(nWk), 256, 0, stream>>>(Wv, Wvb, nWk);
  cast_bf16_kernel<<<blocks4(nWo), 256, 0, stream>>>(Wo, Wob, nWo);

  // QKV projections
  gemm_nt_kernel<false><<<dim3(QD_ / 128, MT_ / 128), 256, 0, stream>>>(Xb, Wqb, Qb, nullptr, MT_, QD_, H_);
  gemm_nt_kernel<false><<<dim3(KD_ / 128, MT_ / 128), 256, 0, stream>>>(Xb, Wkb, Kb, nullptr, MT_, KD_, H_);
  gemm_nt_kernel<false><<<dim3(KD_ / 128, MT_ / 128), 256, 0, stream>>>(Xb, Wvb, Vb, nullptr, MT_, KD_, H_);

  // RoPE on Q and K
  {
    int tq = MT_ * NH_ * 32;
    int tk = MT_ * NKV_ * 32;
    rope_kernel<<<(tq + 255) / 256, 256, 0, stream>>>(Qb, NH_, tq);
    rope_kernel<<<(tk + 255) / 256, 256, 0, stream>>>(Kb, NKV_, tk);
  }

  // flash attention
  attn_kernel<<<dim3(S_ / 64, NH_, B_), 128, 0, stream>>>(Qb, Kb, Vb, AOb);

  // output projection (f32 out)
  gemm_nt_kernel<true><<<dim3(H_ / 128, MT_ / 128), 256, 0, stream>>>(AOb, Wob, nullptr, out, MT_, H_, QD_);
}
